// LLaMAAttention_91268055040087
// MI455X (gfx1250) — matrix operand order repaired and checked by reading, not compiled, not run on hardware
//
#include <hip/hip_runtime.h>
#include <hip/hip_bf16.h>

#ifndef __has_builtin
#define __has_builtin(x) 0
#endif

#if __has_builtin(__builtin_amdgcn_global_load_async_to_lds_b128)
#define HAVE_ASYNC_LDS 1
#else
#define HAVE_ASYNC_LDS 0
#endif

// ---------------------------------------------------------------------------
// Types for CDNA5 WMMA (wave32): v_wmma_f32_16x16x32_bf16
// ---------------------------------------------------------------------------
typedef __bf16 bf16_t;
typedef __attribute__((ext_vector_type(16))) __bf16 v16bf;
typedef __attribute__((ext_vector_type(8)))  __bf16 v8bf;
typedef __attribute__((ext_vector_type(8)))  float  v8f;

union Frag16 { v16bf v; v8bf h[2]; };

__device__ __forceinline__ v8f wmma_bf16(v16bf a, v16bf b, v8f c) {
  return __builtin_amdgcn_wmma_f32_16x16x32_bf16(false, a, false, b, (short)0, c,
                                                 false, false);
}

#if HAVE_ASYNC_LDS
// Async global->LDS DMA (ASYNCcnt-tracked), b128 per lane.
// Param 1 is <4 x int> addrspace(1)*, param 2 is <4 x int> addrspace(3)*
// (from the round-2 diagnostic).  Cast via integers so the address-space
// conversion always typechecks; AS3 truncation to 32 bits matches the ISA's
// LDS aperture rule (the low 32 bits are the LDS address).
typedef __attribute__((ext_vector_type(4))) int v4i;
typedef __attribute__((address_space(1))) v4i v4i_gl;
typedef __attribute__((address_space(3))) v4i v4i_ds;
__device__ __forceinline__ void async_copy_b128(const void* g, void* l) {
  __builtin_amdgcn_global_load_async_to_lds_b128(
      (v4i_gl*)(unsigned long long)g,
      (v4i_ds*)(unsigned int)(unsigned long long)l, 0, 0);
}
__device__ __forceinline__ void wait_asynccnt0() {
#if __has_builtin(__builtin_amdgcn_s_wait_asynccnt)
  __builtin_amdgcn_s_wait_asynccnt(0);
#else
  asm volatile("s_wait_asynccnt 0x0" ::: "memory");
#endif
}
#endif

// A fragment (16x32 bf16 MxK) from row-major LDS [row][k]:
// lane M = lane%16, half = lane/16; two contiguous 8-elem (16B) runs.
__device__ __forceinline__ v16bf load_a_frag(const bf16_t* base, int stride,
                                             int ks, int lane) {
  int m = lane & 15, hlf = lane >> 4;
  const bf16_t* p = base + (size_t)m * stride + ks + hlf * 8;
  Frag16 f;
  f.h[0] = *(const v8bf*)(p);
  f.h[1] = *(const v8bf*)(p + 16);
  return f.v;
}

// B fragment (32x16 bf16 KxN) from transposed LDS [n][k]:
// lane N = lane%16, khalf = lane/16; two contiguous 8-elem (16B) runs.
__device__ __forceinline__ v16bf load_b_frag(const bf16_t* base, int stride,
                                             int ks, int lane) {
  int n = lane & 15, kh = lane >> 4;
  const bf16_t* p = base + (size_t)n * stride + ks + kh * 8;
  Frag16 f;
  f.h[0] = *(const v8bf*)(p);
  f.h[1] = *(const v8bf*)(p + 16);
  return f.v;
}

// ---------------------------------------------------------------------------
// f32 -> bf16 conversion
// ---------------------------------------------------------------------------
__global__ void cvt_f32_bf16_kernel(const float* __restrict__ x,
                                    bf16_t* __restrict__ y, int n) {
  int i = blockIdx.x * blockDim.x + threadIdx.x;
  if (i < n) y[i] = (bf16_t)x[i];
}

// ---------------------------------------------------------------------------
// WMMA GEMM, double-buffered + software pipelined.
// C[M,N] f32 = A[M,K] bf16 x B[K,N] bf16 (both row-major).
// Block tile 256x128, K-step 64, 256 threads (8 waves), wave tile 64x64
// (16 WMMA per 4 A-frag + 4 B-frag LDS loads).
// Dynamic LDS: 2*(256*64 + 128*64)*2B = 96 KB.
// ---------------------------------------------------------------------------
__global__ __launch_bounds__(256) void gemm_bf16_kernel(
    const bf16_t* __restrict__ A, const bf16_t* __restrict__ B,
    float* __restrict__ C, int M, int N, int K) {
  extern __shared__ char smem[];
  bf16_t* As = (bf16_t*)smem;          // [2][256][64]
  bf16_t* Bt = As + 2 * 256 * 64;      // [2][128][64] transposed [n][k]

  const int t = threadIdx.x;
  const int lane = t & 31, wave = t >> 5;
  const int bm = blockIdx.y * 256, bn = blockIdx.x * 128;
  const int wm = (wave & 3) * 64;   // wave row base
  const int wn = (wave >> 2) * 64;  // wave col base

  v8f acc[4][4];
#pragma unroll
  for (int i = 0; i < 4; ++i)
#pragma unroll
    for (int j = 0; j < 4; ++j)
#pragma unroll
      for (int e = 0; e < 8; ++e) acc[i][j][e] = 0.f;

  v8bf ra[8];  // A staging: 256x64 / 8 / 256thr = 8 vectors
  v8bf rb[4];  // B staging: 64x128 / 8 / 256thr = 4 vectors

  auto gload = [&](int k0) {
#pragma unroll
    for (int i = 0; i < 8; ++i) {
      int v = t + i * 256;
      int r = v >> 3, c8 = (v & 7) * 8;
      ra[i] = *(const v8bf*)(A + (size_t)(bm + r) * K + k0 + c8);
    }
#pragma unroll
    for (int i = 0; i < 4; ++i) {
      int v = t + i * 256;
      int kr = v >> 4, c8 = (v & 15) * 8;
      rb[i] = *(const v8bf*)(B + (size_t)(k0 + kr) * N + bn + c8);
    }
  };
  auto lstore = [&](int buf) {
    bf16_t* as = As + buf * 256 * 64;
    bf16_t* bt = Bt + buf * 128 * 64;
#pragma unroll
    for (int i = 0; i < 8; ++i) {
      int v = t + i * 256;
      int r = v >> 3, c8 = (v & 7) * 8;
      *(v8bf*)(as + r * 64 + c8) = ra[i];
    }
#pragma unroll
    for (int i = 0; i < 4; ++i) {
      int v = t + i * 256;
      int kr = v >> 4, c8 = (v & 15) * 8;
#pragma unroll
      for (int j = 0; j < 8; ++j) bt[(c8 + j) * 64 + kr] = rb[i][j];
    }
  };

  gload(0);
  lstore(0);
  int cur = 0;
  for (int k0 = 0; k0 < K; k0 += 64) {
    __syncthreads();
    const bool more = (k0 + 64 < K);
    if (more) gload(k0 + 64);  // global loads in flight across the WMMA burst
    const bf16_t* as = As + cur * 256 * 64;
    const bf16_t* bt = Bt + cur * 128 * 64;
#pragma unroll
    for (int ks = 0; ks < 64; ks += 32) {
      v16bf a[4];
#pragma unroll
      for (int ti = 0; ti < 4; ++ti)
        a[ti] = load_a_frag(as + (size_t)(wm + ti * 16) * 64, 64, ks, lane);
#pragma unroll
      for (int tj = 0; tj < 4; ++tj) {
        v16bf b = load_b_frag(bt + (size_t)(wn + tj * 16) * 64, 64, ks, lane);
#pragma unroll
        for (int ti = 0; ti < 4; ++ti)
          acc[ti][tj] = wmma_bf16(a[ti], b, acc[ti][tj]);
      }
    }
    if (more) lstore(cur ^ 1);  // s_wait_loadcnt only lands here
    cur ^= 1;
  }

  // C layout per 16x16 tile: VGPR r -> (M = r + 8*(lane/16), N = lane%16)
  const int hlf = lane >> 4, n16 = lane & 15;
#pragma unroll
  for (int ti = 0; ti < 4; ++ti)
#pragma unroll
    for (int tj = 0; tj < 4; ++tj)
#pragma unroll
      for (int r = 0; r < 8; ++r) {
        int row = bm + wm + ti * 16 + r + 8 * hlf;
        int col = bn + wn + tj * 16 + n16;
        C[(size_t)row * N + col] = acc[ti][tj][r];
      }
}

// ---------------------------------------------------------------------------
// RoPE + layout shuffle.  Grid: (2048, 48), block 64.
// ---------------------------------------------------------------------------
__global__ void rope_kernel(const float* __restrict__ Qf,
                            const float* __restrict__ Kf,
                            const float* __restrict__ Vf,
                            bf16_t* __restrict__ Qbf, bf16_t* __restrict__ Kbf,
                            bf16_t* __restrict__ Vbf,
                            float* __restrict__ k_out,
                            float* __restrict__ v_out) {
  const int i = threadIdx.x;   // 0..63 rotation pair index
  const int s = blockIdx.x;    // position
  const int hz = blockIdx.y;   // head slot
  float ang = (float)s * __expf(-((float)(2 * i) / 128.f) * 9.210340371976184f);
  float c = __cosf(ang), sn = __sinf(ang);

  if (hz < 32) {
    int h = hz;
    size_t src = (size_t)s * 4096 + h * 128 + i;
    float x1 = Qf[src], x2 = Qf[src + 64];
    float y1 = x1 * c - x2 * sn;
    float y2 = x2 * c + x1 * sn;
    size_t o = ((size_t)h * 2048 + s) * 128;
    Qbf[o + i] = (bf16_t)y1;
    Qbf[o + i + 64] = (bf16_t)y2;
  } else if (hz < 40) {
    int kh = hz - 32;
    size_t src = (size_t)s * 1024 + kh * 128 + i;
    float x1 = Kf[src], x2 = Kf[src + 64];
    float y1 = x1 * c - x2 * sn;
    float y2 = x2 * c + x1 * sn;
    size_t o = ((size_t)kh * 2048 + s) * 128;
    Kbf[o + i] = (bf16_t)y1;
    Kbf[o + i + 64] = (bf16_t)y2;
#pragma unroll
    for (int r = 0; r < 4; ++r) {
      size_t oo = ((size_t)(kh * 4 + r) * 2048 + s) * 128;
      k_out[oo + i] = y1;
      k_out[oo + i + 64] = y2;
    }
  } else {
    int kh = hz - 40;
    size_t src = (size_t)s * 1024 + kh * 128 + i;
    float x1 = Vf[src], x2 = Vf[src + 64];
    size_t o = ((size_t)kh * 2048 + s) * 128;
    Vbf[o + i] = (bf16_t)x1;
    Vbf[o + i + 64] = (bf16_t)x2;
#pragma unroll
    for (int r = 0; r < 4; ++r) {
      size_t oo = ((size_t)(kh * 4 + r) * 2048 + s) * 128;
      v_out[oo + i] = x1;
      v_out[oo + i + 64] = x2;
    }
  }
}

// ---------------------------------------------------------------------------
// Flash attention (causal, GQA 4:1).  Grid: (16 q-blocks, 32 heads), 256 thr.
// Dynamic LDS: Qs, Ks, Vt(transposed), Ps = 128 KB.
// K tile uses the gfx1250 async-to-LDS path; next K/V block is prefetched
// (global_prefetch_b8) while this one is consumed.
// ---------------------------------------------------------------------------
__global__ __launch_bounds__(256) void flash_attn_kernel(
    const bf16_t* __restrict__ Qh, const bf16_t* __restrict__ Kh,
    const bf16_t* __restrict__ Vh, bf16_t* __restrict__ attn_out) {
  extern __shared__ char smem[];
  bf16_t* Qs = (bf16_t*)smem;          // [128][128]
  bf16_t* Ks = Qs + 128 * 128;         // [128][128] k-row major
  bf16_t* Vt = Ks + 128 * 128;         // [128][128] transposed [d][kv]
  bf16_t* Ps = Vt + 128 * 128;         // [128][128] P round-trip

  const int t = threadIdx.x;
  const int lane = t & 31, wave = t >> 5;
  const int qb = blockIdx.x, head = blockIdx.y;
  const int kvh = head >> 2;
  const int hlf = lane >> 4, l16 = lane & 15;

  const bf16_t* Qg = Qh + ((size_t)head * 2048 + qb * 128) * 128;
  const bf16_t* Kg = Kh + (size_t)kvh * 2048 * 128;
  const bf16_t* Vg = Vh + (size_t)kvh * 2048 * 128;

  // Load Q block (128x128)
#pragma unroll
  for (int i = 0; i < 8; ++i) {
    int v = t + i * 256;
    int r = v >> 4, c8 = (v & 15) * 8;
    *(v8bf*)&Qs[r * 128 + c8] = *(const v8bf*)(Qg + r * 128 + c8);
  }

  v8f O[8];
  float mrow[8], lrow[8];
#pragma unroll
  for (int d = 0; d < 8; ++d)
#pragma unroll
    for (int e = 0; e < 8; ++e) O[d][e] = 0.f;
#pragma unroll
  for (int r = 0; r < 8; ++r) { mrow[r] = -1e30f; lrow[r] = 0.f; }

  const float scale = 0.08838834764831845f;  // 1/sqrt(128)
  const int qrow_base = qb * 128 + wave * 16;

  for (int kb = 0; kb <= qb; ++kb) {
    __syncthreads();  // previous iteration's LDS reads complete

    // K tile (row-major copy): async DMA to LDS when available
#if HAVE_ASYNC_LDS
#pragma unroll
    for (int i = 0; i < 8; ++i) {
      int v = t + i * 256;
      int r = v >> 4, c8 = (v & 15) * 8;
      async_copy_b128(Kg + (size_t)(kb * 128 + r) * 128 + c8,
                      Ks + r * 128 + c8);
    }
#else
#pragma unroll
    for (int i = 0; i < 8; ++i) {
      int v = t + i * 256;
      int r = v >> 4, c8 = (v & 15) * 8;
      *(v8bf*)&Ks[r * 128 + c8] =
          *(const v8bf*)(Kg + (size_t)(kb * 128 + r) * 128 + c8);
    }
#endif
    // V tile, transposed into Vt[d][kv] (register transpose)
#pragma unroll
    for (int i = 0; i < 8; ++i) {
      int v = t + i * 256;
      int r = v >> 4, c8 = (v & 15) * 8;
      v8bf tv = *(const v8bf*)(Vg + (size_t)(kb * 128 + r) * 128 + c8);
#pragma unroll
      for (int j = 0; j < 8; ++j) Vt[(c8 + j) * 128 + r] = tv[j];
    }
    // Prefetch next K/V block: 256 threads x one 128B line = 32KB tile each
    if (kb < qb) {
      const size_t nxt = (size_t)((kb + 1) * 128) * 128 + (size_t)t * 64;
      __builtin_prefetch(Kg + nxt, 0, 1);
      __builtin_prefetch(Vg + nxt, 0, 1);
    }
#if HAVE_ASYNC_LDS
    wait_asynccnt0();
#endif
    __syncthreads();

    // Scores: S(16x128) = Qwave(16x128) x K(128x128)^T
    v8f S[8];
#pragma unroll
    for (int n = 0; n < 8; ++n)
#pragma unroll
      for (int e = 0; e < 8; ++e) S[n][e] = 0.f;
#pragma unroll
    for (int ks = 0; ks < 128; ks += 32) {
      v16bf a = load_a_frag(Qs + (size_t)(wave * 16) * 128, 128, ks, lane);
#pragma unroll
      for (int n = 0; n < 8; ++n) {
        v16bf b = load_b_frag(Ks + (size_t)(n * 16) * 128, 128, ks, lane);
        S[n] = wmma_bf16(a, b, S[n]);
      }
    }

    // Scale + causal mask (diagonal block only)
    const bool diag = (kb == qb);
#pragma unroll
    for (int n = 0; n < 8; ++n)
#pragma unroll
      for (int r = 0; r < 8; ++r) {
        float sv = S[n][r] * scale;
        if (diag) {
          int q = qrow_base + r + 8 * hlf;
          int k = kb * 128 + n * 16 + l16;
          if (k > q) sv = -1e30f;
        }
        S[n][r] = sv;
      }

    // Online softmax per row (C layout: row = r + 8*(lane/16), N = lane%16)
#pragma unroll
    for (int r = 0; r < 8; ++r) {
      float mx = -1e30f;
#pragma unroll
      for (int n = 0; n < 8; ++n) mx = fmaxf(mx, S[n][r]);
#pragma unroll
      for (int msk = 1; msk < 16; msk <<= 1)
        mx = fmaxf(mx, __shfl_xor(mx, msk, 32));
      float mnew = fmaxf(mrow[r], mx);
      float alpha = __expf(mrow[r] - mnew);
      float sum = 0.f;
#pragma unroll
      for (int n = 0; n < 8; ++n) {
        float p = __expf(S[n][r] - mnew);
        S[n][r] = p;
        sum += p;
      }
#pragma unroll
      for (int msk = 1; msk < 16; msk <<= 1) sum += __shfl_xor(sum, msk, 32);
      lrow[r] = lrow[r] * alpha + sum;
      mrow[r] = mnew;
#pragma unroll
      for (int d = 0; d < 8; ++d) O[d][r] *= alpha;
    }

    // P: C layout -> LDS -> A layout (wave-private 16 rows of Ps)
#pragma unroll
    for (int n = 0; n < 8; ++n)
#pragma unroll
      for (int r = 0; r < 8; ++r)
        Ps[(size_t)(wave * 16 + r + 8 * hlf) * 128 + n * 16 + l16] =
            (bf16_t)S[n][r];
    __syncthreads();

    // O(16x128) += P(16x128) x V(128x128)
#pragma unroll
    for (int ks = 0; ks < 128; ks += 32) {
      v16bf a = load_a_frag(Ps + (size_t)(wave * 16) * 128, 128, ks, lane);
#pragma unroll
      for (int d = 0; d < 8; ++d) {
        v16bf b = load_b_frag(Vt + (size_t)(d * 16) * 128, 128, ks, lane);
        O[d] = wmma_bf16(a, b, O[d]);
      }
    }
  }

  // Normalize and write attn output as bf16 [s][head*128 + d]
#pragma unroll
  for (int r = 0; r < 8; ++r) {
    float inv = 1.f / lrow[r];
    int srow = qb * 128 + wave * 16 + r + 8 * hlf;
#pragma unroll
    for (int d = 0; d < 8; ++d)
      attn_out[(size_t)srow * 4096 + head * 128 + d * 16 + l16] =
          (bf16_t)(O[d][r] * inv);
  }
}

// ---------------------------------------------------------------------------
// Host-side orchestration
// ---------------------------------------------------------------------------
extern "C" void kernel_launch(void* const* d_in, const int* in_sizes, int n_in,
                              void* d_out, int out_size, void* d_ws,
                              size_t ws_size, hipStream_t stream) {
  (void)in_sizes; (void)n_in; (void)out_size; (void)ws_size;
  const float* hidden = (const float*)d_in[0];  // [2048,4096]
  const float* Wq = (const float*)d_in[1];      // [4096,4096]
  const float* Wk = (const float*)d_in[2];      // [4096,1024]
  const float* Wv = (const float*)d_in[3];      // [4096,1024]
  const float* Wo = (const float*)d_in[4];      // [4096,4096]

  const int S = 2048, H = 4096, KV = 1024;

  // d_out: out [2048*4096] | k [32*2048*128] | v [32*2048*128], all f32
  float* out_p = (float*)d_out;
  float* k_out = out_p + (size_t)S * H;
  float* v_out = k_out + (size_t)S * H;

  // Workspace carve (bytes)
  char* ws = (char*)d_ws;
  bf16_t* hbf  = (bf16_t*)(ws);                         // 16 MB
  bf16_t* Wqb  = (bf16_t*)(ws + 16777216ull);           // 32 MB
  bf16_t* Wkb  = (bf16_t*)(ws + 50331648ull);           // 8 MB
  bf16_t* Wvb  = (bf16_t*)(ws + 58720256ull);           // 8 MB
  bf16_t* Wob  = (bf16_t*)(ws + 67108864ull);           // 32 MB
  float*  Qf   = (float*) (ws + 100663296ull);          // 32 MB
  float*  Kf   = (float*) (ws + 134217728ull);          // 8 MB
  float*  Vf   = (float*) (ws + 142606336ull);          // 8 MB
  bf16_t* Qbf  = (bf16_t*)(ws + 150994944ull);          // 16 MB
  bf16_t* Kbf  = (bf16_t*)(ws + 167772160ull);          // 4 MB
  bf16_t* Vbf  = (bf16_t*)(ws + 171966464ull);          // 4 MB
  bf16_t* Abf  = (bf16_t*)(ws + 176160768ull);          // 16 MB (attn out)

  auto cvt = [&](const float* src, bf16_t* dst, int n) {
    cvt_f32_bf16_kernel<<<(n + 255) / 256, 256, 0, stream>>>(src, dst, n);
  };
  cvt(hidden, hbf, S * H);
  cvt(Wq, Wqb, H * H);
  cvt(Wk, Wkb, H * KV);
  cvt(Wv, Wvb, H * KV);
  cvt(Wo, Wob, H * H);

  const size_t gemm_lds = 2 * (256 * 64 + 128 * 64) * sizeof(bf16_t);  // 96 KB

  // Projections (grid: N/128 x M/256)
  gemm_bf16_kernel<<<dim3(H / 128, S / 256), 256, gemm_lds, stream>>>(
      hbf, Wqb, Qf, S, H, H);
  gemm_bf16_kernel<<<dim3(KV / 128, S / 256), 256, gemm_lds, stream>>>(
      hbf, Wkb, Kf, S, KV, H);
  gemm_bf16_kernel<<<dim3(KV / 128, S / 256), 256, gemm_lds, stream>>>(
      hbf, Wvb, Vf, S, KV, H);

  // RoPE + per-head layouts + f32 repeated K/V outputs
  rope_kernel<<<dim3(S, 48), 64, 0, stream>>>(Qf, Kf, Vf, Qbf, Kbf, Vbf,
                                              k_out, v_out);

  // Flash attention (128 KB dynamic LDS)
  flash_attn_kernel<<<dim3(S / 128, 32), 256, 131072, stream>>>(Qbf, Kbf, Vbf,
                                                                Abf);

  // Output projection
  gemm_bf16_kernel<<<dim3(H / 128, S / 256), 256, gemm_lds, stream>>>(
      Abf, Wob, out_p, S, H, H);
}
